// UserBehaviorGNN_4277787427471
// MI455X (gfx1250) — matrix operand order repaired and checked by reading, not compiled, not run on hardware
//
#include <hip/hip_runtime.h>
#include <hip/hip_bf16.h>

#define CDIM 128
#define HH 8
#define DD 16
#define LL 2

typedef __attribute__((ext_vector_type(16))) __bf16 v16bf;
typedef __attribute__((ext_vector_type(2)))  __bf16 bf16x2;
typedef __attribute__((ext_vector_type(8)))  float  v8f;

__device__ __forceinline__ float fast_sigmoid(float t) {
  // v_exp_f32 + v_add + v_rcp_f32 (no IEEE-division expansion)
  return __builtin_amdgcn_rcpf(1.0f + __expf(-t));
}

// tanh-GELU via sigmoid identity: 0.5*(1+tanh(u)) == sigmoid(2u). Branch-free.
__device__ __forceinline__ float gelu_fast(float x) {
  float t = 1.5957691216057308f * (x + 0.044715f * x * x * x);
  return x * fast_sigmoid(t);
}

// order-preserving float<->uint mapping for atomic max on signed floats
__device__ __forceinline__ unsigned f2sort(float x) {
  unsigned u = __float_as_uint(x);
  return (u & 0x80000000u) ? ~u : (u | 0x80000000u);
}
__device__ __forceinline__ float sort2f(unsigned k) {
  return __uint_as_float((k & 0x80000000u) ? (k & 0x7fffffffu) : ~k);
}

// ---------------- embeddings: relu(x[N,3] @ W[3,128] + b) ----------------
__global__ void k_embed(const float* __restrict__ xm, const float* __restrict__ xc,
                        const float* __restrict__ Wemb, const float* __restrict__ bemb,
                        float* __restrict__ xout, int N) {
  long idx = (long)blockIdx.x * blockDim.x + threadIdx.x;
  long total = 2L * N * CDIM;
  if (idx >= total) return;
  int t = (int)(idx / ((long)N * CDIM));
  long rem = idx - (long)t * N * CDIM;
  int n = (int)(rem / CDIM);
  int c = (int)(rem % CDIM);
  const float* x = (t == 0) ? xm : xc;
  const float* W = Wemb + (size_t)t * 3 * CDIM;
  float acc = bemb[t * CDIM + c];
  acc += x[n * 3 + 0] * W[0 * CDIM + c];
  acc += x[n * 3 + 1] * W[1 * CDIM + c];
  acc += x[n * 3 + 2] * W[2 * CDIM + c];
  xout[(size_t)t * N * CDIM + (size_t)n * CDIM + c] = fmaxf(acc, 0.f);
}

// ---------------- weight prep ----------------
// W is [in=128][out=128] f32 (x @ W layout). Wt is [out][in] bf16. Optional bias copy.
__global__ void k_wt_bf16(const float* __restrict__ W, const float* __restrict__ b,
                          __bf16* __restrict__ Wt, float* __restrict__ bout) {
  int idx = blockIdx.x * blockDim.x + threadIdx.x;  // 16384
  int o = idx >> 7, i = idx & 127;
  Wt[idx] = (__bf16)W[i * CDIM + o];
  if (bout && i == 0) bout[o] = b[o];
}

// effective relation weight: Wt_eff[o=h*16+e][i] = sum_d W[i][h*16+d]*rel[h][d][e]
// and beff[o] = sum_d b[h*16+d]*rel[h][d][e]
__global__ void k_weff(const float* __restrict__ W, const float* __restrict__ rel,
                       const float* __restrict__ b, __bf16* __restrict__ Wt,
                       float* __restrict__ beff) {
  int idx = blockIdx.x * blockDim.x + threadIdx.x;  // 16384
  int o = idx >> 7, i = idx & 127;
  int h = o >> 4, eo = o & 15;
  const float* rp = rel + h * DD * DD;
  float acc = 0.f;
#pragma unroll
  for (int d = 0; d < DD; ++d)
    acc += W[i * CDIM + h * DD + d] * rp[d * DD + eo];
  Wt[o * CDIM + i] = (__bf16)acc;
  if (i == 0) {
    float bacc = 0.f;
#pragma unroll
    for (int d = 0; d < DD; ++d) bacc += b[h * DD + d] * rp[d * DD + eo];
    beff[o] = bacc;
  }
}

__global__ void k_prel(const float* __restrict__ p, float* __restrict__ ps) {
  int i = threadIdx.x;
  if (i < 2 * HH) ps[i] = p[i] * 0.25f;  // 1/sqrt(D), D=16
}

__global__ void k_init_stats(unsigned* __restrict__ dmax, float* __restrict__ dsum, int n) {
  int i = blockIdx.x * blockDim.x + threadIdx.x;
  if (i < n) { dmax[i] = f2sort(-3.0e38f); dsum[i] = 0.f; }
}

// ---------------- WMMA GEMM ----------------
// Out[:, local cols] = act(A[nrows,128]) @ Wt[ocols][128]^T + bias, with the output
// column space possibly spanning 3 concatenated 128-col destinations (fused q/k/v).
// One wave computes a 16x64 tile; all 4 B fragments of a K-chunk are preloaded so the
// load clause overlaps the 4 back-to-back v_wmma_f32_16x16x32_bf16.
// GELU / SKIP are compile-time so the hot loop carries no uniform branch chains.
template <bool GELU, bool SKIP>
__global__ __launch_bounds__(32) void k_gemm_wmma(
    const float* __restrict__ A, const __bf16* __restrict__ Wt,
    const float* __restrict__ bias, const float* __restrict__ Xold,
    const float* __restrict__ skipPtr,
    float* __restrict__ Out0, float* __restrict__ Out1, float* __restrict__ Out2,
    int nrows) {
  const int ln = threadIdx.x & 31;
  const int m0 = blockIdx.x << 4;
  const int og0 = blockIdx.y << 6;           // global output-column base (0..383)
  const int sel = og0 >> 7;                  // which destination buffer
  float* __restrict__ Out = (sel == 0) ? Out0 : (sel == 1) ? Out1 : Out2;
  const int nl0 = og0 & 127;                 // column base within destination
  const int mr = ln & 15;
  const int half = ln >> 4;
  const bool fullTile = (m0 + 16 <= nrows);
  int ar = m0 + mr;
  if (ar >= nrows) ar = 0;                   // clamp: rows are independent in A@B
  const float* __restrict__ arow = A + (size_t)ar * CDIM + (half << 4);
  const __bf16* __restrict__ bbase = Wt + (size_t)(og0 + mr) * CDIM + (half << 4);
  __builtin_prefetch(arow, 0, 0);            // global_prefetch_b8

  v8f acc[4];
  acc[0] = v8f{}; acc[1] = v8f{}; acc[2] = v8f{}; acc[3] = v8f{};

#pragma unroll
  for (int kk = 0; kk < CDIM; kk += 32) {
    // A-matrix 16x32 bf16 fragment (VGPR e / lane-half -> K block), immediate offsets
    v16bf a;
#pragma unroll
    for (int e = 0; e < 8; ++e) {
      float2 av = *(const float2*)(arow + kk + (e << 1));
      if (GELU) { av.x = gelu_fast(av.x); av.y = gelu_fast(av.y); }
      a[2 * e]     = (__bf16)av.x;
      a[2 * e + 1] = (__bf16)av.y;
    }
    // preload all 4 B fragments (constant offsets off one base pointer)
    v16bf bfr[4];
#pragma unroll
    for (int c = 0; c < 4; ++c) {
#pragma unroll
      for (int e = 0; e < 8; ++e) {
        bf16x2 p = *(const bf16x2*)(bbase + c * (16 * CDIM) + kk + (e << 1));
        bfr[c][2 * e]     = p[0];
        bfr[c][2 * e + 1] = p[1];
      }
    }
#pragma unroll
    for (int c = 0; c < 4; ++c)
      acc[c] = __builtin_amdgcn_wmma_f32_16x16x32_bf16(false, a, false, bfr[c], (short)0,
                                                       acc[c], false, false);
  }

  float beta = 0.f;
  if (SKIP) beta = fast_sigmoid(skipPtr[0]);

  if (fullTile) {
#pragma unroll
    for (int c = 0; c < 4; ++c) {
      int gcol = og0 + (c << 4) + mr;
      int lcol = nl0 + (c << 4) + mr;
      float bv = bias[gcol];
#pragma unroll
      for (int r = 0; r < 8; ++r) {
        int row = m0 + r + (half << 3);
        float o = acc[c][r] + bv;
        if (SKIP) {
          float xo = Xold[(size_t)row * CDIM + lcol];
          o = fmaxf(beta * o + (1.f - beta) * xo, 0.f);
        }
        Out[(size_t)row * CDIM + lcol] = o;
      }
    }
  } else {
#pragma unroll
    for (int c = 0; c < 4; ++c) {
      int gcol = og0 + (c << 4) + mr;
      int lcol = nl0 + (c << 4) + mr;
      float bv = bias[gcol];
#pragma unroll
      for (int r = 0; r < 8; ++r) {
        int row = m0 + r + (half << 3);
        if (row >= nrows) continue;
        float o = acc[c][r] + bv;
        if (SKIP) {
          float xo = Xold[(size_t)row * CDIM + lcol];
          o = fmaxf(beta * o + (1.f - beta) * xo, 0.f);
        }
        Out[(size_t)row * CDIM + lcol] = o;
      }
    }
  }
}

// ---------------- edge phase ----------------
__global__ void k_edge_alpha(const int* __restrict__ si, const int* __restrict__ di,
                             const float* __restrict__ Q, const float* __restrict__ KRr,
                             const float* __restrict__ ps, float* __restrict__ alpha,
                             unsigned* __restrict__ dmax, int E) {
  long idx = (long)blockIdx.x * blockDim.x + threadIdx.x;
  if (idx >= (long)E * HH) return;
  int e = (int)(idx >> 3), h = (int)(idx & 7);
  int s = si[e], d = di[e];
  const float4* qp = (const float4*)(Q + (size_t)d * CDIM + h * DD);
  const float4* kp = (const float4*)(KRr + (size_t)s * CDIM + h * DD);
  float acc = 0.f;
#pragma unroll
  for (int t = 0; t < 4; ++t) {
    float4 qv = qp[t], kv = kp[t];
    acc += qv.x * kv.x + qv.y * kv.y + qv.z * kv.z + qv.w * kv.w;
  }
  acc *= ps[h];
  alpha[idx] = acc;
  atomicMax(dmax + (size_t)d * HH + h, f2sort(acc));
}

__global__ void k_edge_exp(const int* __restrict__ di, float* __restrict__ alpha,
                           const unsigned* __restrict__ dmax, float* __restrict__ dsum, int E) {
  long idx = (long)blockIdx.x * blockDim.x + threadIdx.x;
  if (idx >= (long)E * HH) return;
  int e = (int)(idx >> 3), h = (int)(idx & 7);
  int d = di[e];
  float m = sort2f(dmax[(size_t)d * HH + h]);
  float ex = __expf(alpha[idx] - m);
  alpha[idx] = ex;
  atomicAdd(dsum + (size_t)d * HH + h, ex);
}

__global__ void k_edge_scatter(const int* __restrict__ si, const int* __restrict__ di,
                               const float* __restrict__ alpha, const float* __restrict__ dsum,
                               const float* __restrict__ VRr, float* __restrict__ agg, int E) {
  long idx = (long)blockIdx.x * blockDim.x + threadIdx.x;
  if (idx >= (long)E * HH) return;
  int e = (int)(idx >> 3), h = (int)(idx & 7);
  int s = si[e], d = di[e];
  float w = alpha[idx] / (dsum[(size_t)d * HH + h] + 1e-16f);
  const float4* vp = (const float4*)(VRr + (size_t)s * CDIM + h * DD);
  float* ap = agg + (size_t)d * CDIM + h * DD;
#pragma unroll
  for (int t = 0; t < 4; ++t) {
    float4 vv = vp[t];
    atomicAdd(ap + 4 * t + 0, w * vv.x);
    atomicAdd(ap + 4 * t + 1, w * vv.y);
    atomicAdd(ap + 4 * t + 2, w * vv.z);
    atomicAdd(ap + 4 * t + 3, w * vv.w);
  }
}

// ---------------- host ----------------
extern "C" void kernel_launch(void* const* d_in, const int* in_sizes, int n_in,
                              void* d_out, int out_size, void* d_ws, size_t ws_size,
                              hipStream_t stream) {
  const float* x_music = (const float*)d_in[0];
  const float* x_cal   = (const float*)d_in[1];
  const int*   edge_mc = (const int*)d_in[2];
  const int*   edge_cm = (const int*)d_in[3];
  const float* W_emb   = (const float*)d_in[4];
  const float* b_emb   = (const float*)d_in[5];
  const float* Wk      = (const float*)d_in[6];
  const float* bk      = (const float*)d_in[7];
  const float* Wq      = (const float*)d_in[8];
  const float* bq      = (const float*)d_in[9];
  const float* Wv      = (const float*)d_in[10];
  const float* bv      = (const float*)d_in[11];
  const float* a_rel   = (const float*)d_in[12];
  const float* m_rel   = (const float*)d_in[13];
  const float* p_rel   = (const float*)d_in[14];
  const float* Wa      = (const float*)d_in[15];
  const float* ba      = (const float*)d_in[16];
  const float* skip    = (const float*)d_in[17];

  const int N = in_sizes[0] / 3;
  const int E = in_sizes[2] / 2;
  const size_t WB = (size_t)CDIM * CDIM;  // 16384 elements per 128x128 weight

  char* ws = (char*)d_ws;
  const size_t nodeBytes = (size_t)2 * N * CDIM * sizeof(float);
  float* xA = (float*)ws; ws += nodeBytes;   // x buffer A
  float* xB = (float*)ws; ws += nodeBytes;   // x buffer B
  float* KR = (float*)ws; ws += nodeBytes;   // k_rel per relation
  float* VR = (float*)ws; ws += nodeBytes;   // v_rel per relation
  float* AG = (float*)ws; ws += nodeBytes;   // aggregated messages per type
  float* ALPHA = (float*)ws; ws += (size_t)E * HH * sizeof(float);
  unsigned* DMAX = (unsigned*)ws; ws += (size_t)N * HH * sizeof(unsigned);
  float* DSUM = (float*)ws; ws += (size_t)N * HH * sizeof(float);
  __bf16* W3  = (__bf16*)ws; ws += (size_t)2 * 3 * WB * sizeof(__bf16);  // [t][Q|Ke|Ve]
  __bf16* WAt = (__bf16*)ws; ws += (size_t)2 * WB * sizeof(__bf16);
  float* B3 = (float*)ws; ws += (size_t)2 * 3 * CDIM * sizeof(float);    // [t][384]
  float* PS = (float*)ws; ws += (size_t)2 * HH * sizeof(float);

  // embeddings
  {
    long tot = 2L * N * CDIM;
    k_embed<<<(unsigned)((tot + 255) / 256), 256, 0, stream>>>(x_music, x_cal, W_emb, b_emb, xA, N);
  }

  dim3 gqkv((N + 15) / 16, 6);   // 384 output cols: Q | K_rel | V_rel
  dim3 gout((N + 15) / 16, 2);   // 128 output cols
  float* xcur = xA;
  for (int l = 0; l < LL; ++l) {
    float* Qb    = (l == 0) ? (float*)d_out : xA;  // reuse dead buffers for Q
    float* xnext = (l == 0) ? xB : (float*)d_out;

    // weight prep (tiny). relation r == source type t for this graph.
    for (int t = 0; t < 2; ++t) {
      __bf16* w3t = W3 + (size_t)t * 3 * WB;
      float* b3t = B3 + (size_t)t * 3 * CDIM;
      k_wt_bf16<<<64, 256, 0, stream>>>(Wq + (size_t)(l * 2 + t) * WB,
                                        bq + (size_t)(l * 2 + t) * CDIM, w3t, b3t);
      k_weff<<<64, 256, 0, stream>>>(Wk + (size_t)(l * 2 + t) * WB,
                                     a_rel + (size_t)(l * 2 + t) * HH * DD * DD,
                                     bk + (size_t)(l * 2 + t) * CDIM,
                                     w3t + WB, b3t + CDIM);
      k_weff<<<64, 256, 0, stream>>>(Wv + (size_t)(l * 2 + t) * WB,
                                     m_rel + (size_t)(l * 2 + t) * HH * DD * DD,
                                     bv + (size_t)(l * 2 + t) * CDIM,
                                     w3t + 2 * WB, b3t + 2 * CDIM);
      k_wt_bf16<<<64, 256, 0, stream>>>(Wa + (size_t)(l * 2 + t) * WB,
                                        nullptr, WAt + (size_t)t * WB, nullptr);
    }
    k_prel<<<1, 32, 0, stream>>>(p_rel + (size_t)l * 2 * HH, PS);

    // fused Q / K_rel / V_rel projection (one pass over xs[t])
    for (int t = 0; t < 2; ++t) {
      k_gemm_wmma<false, false><<<gqkv, 32, 0, stream>>>(
          xcur + (size_t)t * N * CDIM,
          W3 + (size_t)t * 3 * WB,
          B3 + (size_t)t * 3 * CDIM,
          nullptr, nullptr,
          Qb + (size_t)t * N * CDIM,
          KR + (size_t)t * N * CDIM,
          VR + (size_t)t * N * CDIM, N);
    }

    hipMemsetAsync(AG, 0, nodeBytes, stream);
    long eh = (long)E * HH;
    unsigned eblocks = (unsigned)((eh + 255) / 256);
    for (int r = 0; r < 2; ++r) {
      int dt = 1 - r;
      const int* edges = (r == 0) ? edge_mc : edge_cm;
      const int* si = edges;
      const int* di = edges + E;
      k_init_stats<<<(N * HH + 255) / 256, 256, 0, stream>>>(DMAX, DSUM, N * HH);
      k_edge_alpha<<<eblocks, 256, 0, stream>>>(si, di, Qb + (size_t)dt * N * CDIM,
                                                KR + (size_t)r * N * CDIM, PS + r * HH,
                                                ALPHA, DMAX, E);
      k_edge_exp<<<eblocks, 256, 0, stream>>>(di, ALPHA, DMAX, DSUM, E);
      k_edge_scatter<<<eblocks, 256, 0, stream>>>(si, di, ALPHA, DSUM,
                                                  VR + (size_t)r * N * CDIM,
                                                  AG + (size_t)dt * N * CDIM, E);
    }

    // output: relu(beta*(gelu(agg)@Wa + ba) + (1-beta)*x)  — gelu + skip fused in GEMM
    for (int t = 0; t < 2; ++t) {
      float* ot = xnext + (size_t)t * N * CDIM;
      k_gemm_wmma<true, true><<<gout, 32, 0, stream>>>(
          AG + (size_t)t * N * CDIM,
          WAt + (size_t)t * WB,
          ba + (size_t)(l * 2 + t) * CDIM,
          xcur + (size_t)t * N * CDIM,
          skip + (l * 2 + t),
          ot, ot, ot, N);
    }
    xcur = xnext;
  }
}
